// VectorQuantizer_52106543235260
// MI455X (gfx1250) — matrix operand order repaired and checked by reading, not compiled, not run on hardware
//
#include <hip/hip_runtime.h>
#include <hip/hip_bf16.h>
#include <math.h>

typedef _Float16 v8h  __attribute__((ext_vector_type(8)));
typedef _Float16 v16h __attribute__((ext_vector_type(16)));
typedef float    v8f  __attribute__((ext_vector_type(8)));

#define D               256
#define KCB             1024
#define WAVES_PER_BLOCK 2
#define TOKENS_PER_WAVE 32          // two 16-row WMMA tiles per wave
#define TOKENS_PER_BLOCK (WAVES_PER_BLOCK * TOKENS_PER_WAVE)

// ---------------------------------------------------------------------------
// Kernel 1: normalize codebook rows to f16 (WMMA B operand), zero accums.
// ---------------------------------------------------------------------------
__global__ void vq_norm_codebook(const float* __restrict__ embed,
                                 _Float16* __restrict__ cn,
                                 float* __restrict__ counts,
                                 float* __restrict__ sumsq) {
    const int b    = blockIdx.x;
    const int lane = threadIdx.x;          // 0..31, wave32
    const float* row = embed + (size_t)b * D + lane * 8;
    float v[8];
    float ss = 0.0f;
#pragma unroll
    for (int i = 0; i < 8; ++i) { v[i] = row[i]; ss += v[i] * v[i]; }
#pragma unroll
    for (int off = 16; off >= 1; off >>= 1) ss += __shfl_xor(ss, off, 32);
    const float rn = 1.0f / fmaxf(sqrtf(ss), 1e-12f);
    v8h h;
#pragma unroll
    for (int i = 0; i < 8; ++i) h[i] = (_Float16)(v[i] * rn);
    *(v8h*)(cn + (size_t)b * D + lane * 8) = h;

    if (lane == 0) counts[b] = 0.0f;
    if (b == 0 && lane == 1) *sumsq = 0.0f;
}

// ---------------------------------------------------------------------------
// helpers (fully inlined; constant indices keep arrays in VGPRs)
// ---------------------------------------------------------------------------
__device__ __forceinline__ void load_b4(v16h b[4], const _Float16* __restrict__ bp) {
#pragma unroll
    for (int kk = 0; kk < 4; ++kk)
        b[kk] = *(const v16h*)(bp + kk * 32);
}

__device__ __forceinline__ v8f wmma4(const v16h* a, const v16h b[4], v8f acc) {
#pragma unroll
    for (int kk = 0; kk < 4; ++kk)
        acc = __builtin_amdgcn_wmma_f32_16x16x32_f16(
                  false, a[kk], false, b[kk], (short)0, acc, false, false);
    return acc;
}

// ---------------------------------------------------------------------------
// Kernel 2: fused (normalize-z -> WMMA scores -> argmax -> gather -> MSE).
// 64 threads = 2 waves; each wave owns a 32-token tile (two 16-row M-tiles).
// B is double-buffered at half-chunk (4-fragment) granularity to keep the
// whole working set under 256 VGPRs (no s_set_vgpr_msb churn).
// ---------------------------------------------------------------------------
__global__ void __launch_bounds__(WAVES_PER_BLOCK * 32)
__attribute__((amdgpu_waves_per_eu(4))) vq_main(
    const float* __restrict__ z,
    const float* __restrict__ embed,
    const _Float16* __restrict__ cn,
    float* __restrict__ out_zq,
    int* __restrict__ out_idx,
    float* __restrict__ counts,
    float* __restrict__ sumsq) {

    __shared__ _Float16 sz[WAVES_PER_BLOCK][TOKENS_PER_WAVE][D];  // 32 KB
    __shared__ int      sidx[WAVES_PER_BLOCK][TOKENS_PER_WAVE];

    const int  lane = threadIdx.x & 31;
    const int  w    = threadIdx.x >> 5;
    const long tok0 = ((long)blockIdx.x * WAVES_PER_BLOCK + w) * TOKENS_PER_WAVE;

    // ---- normalize 32 z-rows into LDS as f16 ----
    for (int rr = 0; rr < TOKENS_PER_WAVE; ++rr) {
        const float* zr = z + (tok0 + rr) * D + lane * 8;
        float v[8];
        float ss = 0.0f;
#pragma unroll
        for (int i = 0; i < 8; ++i) { v[i] = zr[i]; ss += v[i] * v[i]; }
#pragma unroll
        for (int off = 16; off >= 1; off >>= 1) ss += __shfl_xor(ss, off, 32);
        const float rn = 1.0f / fmaxf(sqrtf(ss), 1e-12f);
        v8h h;
#pragma unroll
        for (int i = 0; i < 8; ++i) h[i] = (_Float16)(v[i] * rn);
        *(v8h*)&sz[w][rr][lane * 8] = h;
    }
    __syncthreads();

    // ---- build 2x8 resident A fragments (16-bit A 16x32, contiguous k order) ----
    // lane l: row = l%16; contiguous k [k0 + (l/16)*16 .. +15]
    const int g = lane >> 4;   // half-wave group
    const int r = lane & 15;   // row-in-tile (A) / column-in-chunk (B,C,D)
    v16h a[2][8];
#pragma unroll
    for (int t = 0; t < 2; ++t) {
#pragma unroll
        for (int kk = 0; kk < 8; ++kk) {
            const int k0 = kk * 32;
            a[t][kk] = *(const v16h*)&sz[w][t * 16 + r][k0 + g * 16];
        }
    }

    // ---- scan codebook: 16 columns per chunk, half-chunk B double-buffer ----
    float best[2][8];
    int   bidx[2][8];
#pragma unroll
    for (int t = 0; t < 2; ++t)
#pragma unroll
        for (int j = 0; j < 8; ++j) { best[t][j] = -3.0e38f; bidx[t][j] = 0; }

    // B layout: lane n holds column n0+(n%16), contiguous K halves,
    // low-K half-wave (lanes 0-15) vs high-K (16-31) -> offset g*16.
    const _Float16* cn_lane = cn + (size_t)r * D + g * 16;

    v16h b0[4], b1[4];
    load_b4(b0, cn_lane);                                      // chunk 0, dims 0..127
    for (int n0 = 0; n0 < KCB; n0 += 16) {
        load_b4(b1, cn_lane + (size_t)n0 * D + 128);           // this chunk, dims 128..255
        v8f acc0 = wmma4(&a[0][0], b0, (v8f){});
        v8f acc1 = wmma4(&a[1][0], b0, (v8f){});
        if (n0 + 16 < KCB)
            load_b4(b0, cn_lane + (size_t)(n0 + 16) * D);      // next chunk, dims 0..127
        acc0 = wmma4(&a[0][4], b1, acc0);
        acc1 = wmma4(&a[1][4], b1, acc1);

        // C/D layout: VGPR j, lane n -> row (n<16 ? j : j+8), col n0 + (n&15)
        const int cand = n0 + r;
#pragma unroll
        for (int j = 0; j < 8; ++j) {
            if (acc0[j] > best[0][j]) { best[0][j] = acc0[j]; bidx[0][j] = cand; }
            if (acc1[j] > best[1][j]) { best[1][j] = acc1[j]; bidx[1][j] = cand; }
        }
    }

    // ---- argmax reduction across the 16 lanes of each half-wave ----
#pragma unroll
    for (int t = 0; t < 2; ++t) {
#pragma unroll
        for (int j = 0; j < 8; ++j) {
            float v   = best[t][j];
            int   idx = bidx[t][j];
#pragma unroll
            for (int off = 8; off >= 1; off >>= 1) {
                float v2 = __shfl_xor(v, off, 32);
                int   i2 = __shfl_xor(idx, off, 32);
                if (v2 > v || (v2 == v && i2 < idx)) { v = v2; idx = i2; }
            }
            const int m = t * 16 + ((lane < 16) ? j : (j + 8));
            if (r == 0) {              // lane 0 (rows j) and lane 16 (rows j+8)
                sidx[w][m]        = idx;
                out_idx[tok0 + m] = idx;
                atomicAdd(&counts[idx], 1.0f);
            }
        }
    }
    __syncthreads();

    // ---- epilogue: gather fp32 codebook rows, write z_q_st, accumulate MSE ----
    float lss = 0.0f;
    for (int m = 0; m < TOKENS_PER_WAVE; ++m) {
        const int e = sidx[w][m];
        const float* er = embed  + (size_t)e * D + lane * 8;
        const float* zr = z      + (tok0 + m) * D + lane * 8;
        float*       op = out_zq + (tok0 + m) * D + lane * 8;
#pragma unroll
        for (int i = 0; i < 8; ++i) {
            const float q  = er[i];
            const float dz = q - zr[i];
            lss += dz * dz;
            op[i] = q;                 // z + sg(z_q - z) == z_q in forward
        }
    }
#pragma unroll
    for (int off = 16; off >= 1; off >>= 1) lss += __shfl_xor(lss, off, 32);
    if (lane == 0) atomicAdd(sumsq, lss);
}

// ---------------------------------------------------------------------------
// Kernel 3: loss + perplexity (single block of K threads).
// ---------------------------------------------------------------------------
__global__ void vq_finalize(const float* __restrict__ counts,
                            const float* __restrict__ sumsq,
                            float* __restrict__ out_loss,
                            float* __restrict__ out_perp,
                            float inv_ntok, float inv_nelem) {
    __shared__ float red[KCB];
    const int t = threadIdx.x;
    const float p = counts[t] * inv_ntok;
    red[t] = p * logf(p + 1e-10f);
    __syncthreads();
    for (int s = KCB / 2; s >= 1; s >>= 1) {
        if (t < s) red[t] += red[t + s];
        __syncthreads();
    }
    if (t == 0) {
        *out_perp = expf(-red[0]);
        *out_loss = 1.25f * (*sumsq) * inv_nelem;   // (BETA + 1) * MSE
    }
}

// ---------------------------------------------------------------------------
extern "C" void kernel_launch(void* const* d_in, const int* in_sizes, int n_in,
                              void* d_out, int out_size, void* d_ws, size_t ws_size,
                              hipStream_t stream) {
    const float* z     = (const float*)d_in[0];      // [32,4096,256] f32
    const float* embed = (const float*)d_in[1];      // [1024,256]    f32
    const int nz   = in_sizes[0];                    // 33554432
    const int ntok = nz / D;                         // 131072
    const int K    = in_sizes[1] / D;                // 1024

    // workspace: [f16 codebook | counts f32[K] | sumsq f32]
    _Float16* cn     = (_Float16*)d_ws;
    float*    counts = (float*)((char*)d_ws + (size_t)K * D * sizeof(_Float16));
    float*    sumsq  = counts + K;

    // outputs packed in tuple order: z_q_st, loss, perplexity, indices(int32)
    float* out_zq   = (float*)d_out;
    float* out_loss = out_zq + (size_t)nz;
    float* out_perp = out_loss + 1;
    int*   out_idx  = (int*)(out_perp + 1);

    vq_norm_codebook<<<K, 32, 0, stream>>>(embed, cn, counts, sumsq);

    const int blocks = ntok / TOKENS_PER_BLOCK;      // 2048
    vq_main<<<blocks, WAVES_PER_BLOCK * 32, 0, stream>>>(
        z, embed, cn, out_zq, out_idx, counts, sumsq);

    vq_finalize<<<1, K, 0, stream>>>(counts, sumsq, out_loss, out_perp,
                                     1.0f / (float)ntok, 1.0f / (float)nz);
}
